// NativePyTorchINT8Dynamic_61040075211287
// MI455X (gfx1250) — 8-bit matrix operand order repaired and checked by reading, compiled, not run on hardware
//
#include <hip/hip_runtime.h>
#include <cstdint>

typedef __attribute__((ext_vector_type(8))) int v8i;
typedef __attribute__((ext_vector_type(4))) int v4i;

#define BM 128
#define BN 128
#define BK 64
#define LDS_STRIDE 80

#if __has_builtin(__builtin_amdgcn_global_load_async_to_lds_b128)
#define USE_ASYNC_LDS 1
#else
#define USE_ASYNC_LDS 0
#endif

__device__ __forceinline__ void copy16_to_lds(const int8_t* g, int8_t* l) {
#if USE_ASYNC_LDS
  __builtin_amdgcn_global_load_async_to_lds_b128(
      (__attribute__((address_space(1))) v4i*)(g),
      (__attribute__((address_space(3))) v4i*)(l), 0, 0);
#else
  *(int4*)l = *(const int4*)g;
#endif
}

__device__ __forceinline__ void wait_async_copies() {
#if USE_ASYNC_LDS
  asm volatile("s_wait_asynccnt 0" ::: "memory");
#endif
}

__global__ __launch_bounds__(256)
void quant_rows_k4096(const float* __restrict__ src, int8_t* __restrict__ dst,
                      float* __restrict__ scales) {
  const int row = blockIdx.x;
  const int tid = threadIdx.x;
  const float* rp = src + (size_t)row * 4096;

  float4 v[4];
  float mx = 0.0f;
#pragma unroll
  for (int i = 0; i < 4; ++i) {
    v[i] = ((const float4*)rp)[tid + i * 256];
    mx = fmaxf(mx, fmaxf(fmaxf(fabsf(v[i].x), fabsf(v[i].y)),
                         fmaxf(fabsf(v[i].z), fabsf(v[i].w))));
  }
#pragma unroll
  for (int off = 16; off > 0; off >>= 1)
    mx = fmaxf(mx, __shfl_xor(mx, off, 32));

  __shared__ float red[8];
  __shared__ float s_inv;
  if ((tid & 31) == 0) red[tid >> 5] = mx;
  __syncthreads();
  if (tid == 0) {
    float m = red[0];
#pragma unroll
    for (int i = 1; i < 8; ++i) m = fmaxf(m, red[i]);
    const float sc = fmaxf(m, 1e-5f) / 127.0f;
    scales[row] = sc;
    s_inv = 1.0f / sc;
  }
  __syncthreads();
  const float inv = s_inv;

  uint32_t* dp = (uint32_t*)(dst + (size_t)row * 4096);
#pragma unroll
  for (int i = 0; i < 4; ++i) {
    int q0 = (int)rintf(v[i].x * inv);
    int q1 = (int)rintf(v[i].y * inv);
    int q2 = (int)rintf(v[i].z * inv);
    int q3 = (int)rintf(v[i].w * inv);
    q0 = max(-127, min(127, q0));
    q1 = max(-127, min(127, q1));
    q2 = max(-127, min(127, q2));
    q3 = max(-127, min(127, q3));
    uint32_t p = (uint32_t)(q0 & 255) | ((uint32_t)(q1 & 255) << 8) |
                 ((uint32_t)(q2 & 255) << 16) | ((uint32_t)(q3 & 255) << 24);
    dp[tid + i * 256] = p;
  }
}

__global__ __launch_bounds__(256)
void int8_wmma_gemm(const int8_t* __restrict__ xq, const int8_t* __restrict__ wq,
                    const float* __restrict__ xs, const float* __restrict__ wsc,
                    float* __restrict__ out, int M, int N, int K) {
  __shared__ __align__(16) int8_t As[2][BM * LDS_STRIDE];
  __shared__ __align__(16) int8_t Bs[2][BN * LDS_STRIDE];

  const int tid  = threadIdx.x;
  const int lane = tid & 31;
  const int wave = tid >> 5;
  const int wm = (wave & 1) * 64;
  const int wn = (wave >> 1) * 32;
  const int lm = lane & 15;
  const int hi = lane >> 4;

  const int m_base = blockIdx.y * BM;
  const int n_base = blockIdx.x * BN;

  const int row0 = tid >> 2;
  const int col0 = (tid & 3) << 4;
  const int8_t* ag0 = xq + (size_t)(m_base + row0) * K + col0;
  const int8_t* ag1 = xq + (size_t)(m_base + row0 + 64) * K + col0;
  const int8_t* bg0 = wq + (size_t)(n_base + row0) * K + col0;
  const int8_t* bg1 = wq + (size_t)(n_base + row0 + 64) * K + col0;
  const int l0 = row0 * LDS_STRIDE + col0;
  const int l1 = (row0 + 64) * LDS_STRIDE + col0;

  v8i acc[4][2] = {};

  copy16_to_lds(ag0, &As[0][l0]);
  copy16_to_lds(ag1, &As[0][l1]);
  copy16_to_lds(bg0, &Bs[0][l0]);
  copy16_to_lds(bg1, &Bs[0][l1]);
  wait_async_copies();
  __syncthreads();

  int buf = 0;
  for (int ks = 0; ks < K; ks += BK) {
    __builtin_prefetch(ag0 + ks + 2 * BK, 0, 1);
    __builtin_prefetch(bg0 + ks + 2 * BK, 0, 1);

    const int nxt = buf ^ 1;
    if (ks + BK < K) {
      copy16_to_lds(ag0 + ks + BK, &As[nxt][l0]);
      copy16_to_lds(ag1 + ks + BK, &As[nxt][l1]);
      copy16_to_lds(bg0 + ks + BK, &Bs[nxt][l0]);
      copy16_to_lds(bg1 + ks + BK, &Bs[nxt][l1]);
    }

    v8i a[4];
#pragma unroll
    for (int t = 0; t < 4; ++t) {
      const int ro = (wm + t * 16 + lm) * LDS_STRIDE;
#pragma unroll
      for (int v = 0; v < 8; ++v) {
        const int k = ((v >> 1) << 4) + (hi << 3) + ((v & 1) << 2);
        a[t][v] = *(const int*)(&As[buf][ro + k]);
      }
    }
    v8i b[2];
#pragma unroll
    for (int u = 0; u < 2; ++u) {
      const int ro = (wn + u * 16 + lm) * LDS_STRIDE;
#pragma unroll
      for (int v = 0; v < 8; ++v) {
        const int k = ((v >> 1) << 4) + (hi << 3) + ((v & 1) << 2);
        b[u][v] = *(const int*)(&Bs[buf][ro + k]);
      }
    }

#pragma unroll
    for (int t = 0; t < 4; ++t)
#pragma unroll
      for (int u = 0; u < 2; ++u)
        acc[t][u] = __builtin_amdgcn_wmma_i32_16x16x64_iu8(
             true, a[t],  true, b[u], acc[t][u],
             false,  false);

    wait_async_copies();
    __syncthreads();
    buf = nxt;
  }

#pragma unroll
  for (int u = 0; u < 2; ++u) {
    const int gn = n_base + wn + u * 16 + lm;
    const float wsv = wsc[gn];
#pragma unroll
    for (int t = 0; t < 4; ++t) {
#pragma unroll
      for (int r = 0; r < 8; ++r) {
        const int gm = m_base + wm + t * 16 + hi * 8 + r;
        out[(size_t)gm * N + gn] = (float)acc[t][u][r] * xs[gm] * wsv;
      }
    }
  }
}

extern "C" void kernel_launch(void* const* d_in, const int* in_sizes, int n_in,
                              void* d_out, int out_size, void* d_ws, size_t ws_size,
                              hipStream_t stream) {
  const float* x = (const float*)d_in[0];
  const float* w = (const float*)d_in[1];
  float* out = (float*)d_out;

  const int K = 4096;
  const int M = in_sizes[0] / K;
  const int N = in_sizes[1] / K;

  uint8_t* ws = (uint8_t*)d_ws;
  size_t off = 0;
  int8_t* xq  = (int8_t*)(ws + off); off += (size_t)M * K;
  int8_t* wq  = (int8_t*)(ws + off); off += (size_t)N * K;
  float*  xsc = (float*)(ws + off);  off += (size_t)M * sizeof(float);
  float*  wsc = (float*)(ws + off);

  quant_rows_k4096<<<M, 256, 0, stream>>>(x, xq, xsc);
  quant_rows_k4096<<<N, 256, 0, stream>>>(w, wq, wsc);

  dim3 grid(N / BN, M / BM);
  int8_wmma_gemm<<<grid, 256, 0, stream>>>(xq, wq, xsc, wsc, out, M, N, K);
}
